// G2G_LEFTNet_8813272891587
// MI455X (gfx1250) — matrix operand order repaired and checked by reading, not compiled, not run on hardware
//
#include <hip/hip_runtime.h>
#include <stdint.h>

#define HDIM 128
#define NRAD 96
#define H3   384
#define EWD  480   // H3 + NRAD
#define H2   256   // 2*H
#define GGRAPH 64

#define INV_SQRT3 0.57735026918962576f
#define INV_SQRTH 0.08838834764831845f   // 1/sqrt(128)
#define INV_SQRT2 0.70710678118654752f
#define PI_F 3.14159265358979323846f

typedef unsigned short u16;
typedef __attribute__((ext_vector_type(16))) __bf16 v16bf;
typedef __attribute__((ext_vector_type(8)))  float  v8f;

union FragU { unsigned u[8]; v16bf v; };

__device__ __forceinline__ u16 f2bf(float f) {
  union { float f; unsigned u; } c; c.f = f;
  unsigned r = c.u + 0x7fffu + ((c.u >> 16) & 1u);
  return (u16)(r >> 16);
}
__device__ __forceinline__ float siluf(float x) { return x / (1.0f + __expf(-x)); }

// ---------------------------------------------------------------------------
// Weight packer: f32 W[KxN] -> fragment-ready bf16 pairs.
// Fragment tile (ntile, ktile) covers cols [ntile*16,+16), K [ktile*32,+32).
// Packed dword p: tile = p/256 (= ntile*ktiles + ktile), lane = (p/8)%32,
// v = p%8; value packs K = ktile*32 + (lane>>4)*16 + 2v (+1) at col
// ntile*16 + (lane&15). A wave then loads one fragment with 2 b128 loads.
// ---------------------------------------------------------------------------
__global__ __launch_bounds__(256) void pack_weight_kernel(
    const float* __restrict__ W, unsigned* __restrict__ Wp, int K, int N)
{
  int ktiles = K >> 5;
  long total = (long)(N >> 4) * ktiles * 256;
  long p = (long)blockIdx.x * 256 + threadIdx.x;
  if (p >= total) return;
  int v    = (int)(p & 7);
  int lane = (int)((p >> 3) & 31);
  long tile = p >> 8;
  int ktile = (int)(tile % ktiles);
  int ntile = (int)(tile / ktiles);
  int col = ntile * 16 + (lane & 15);
  int k0  = ktile * 32 + (lane >> 4) * 16 + 2 * v;
  unsigned lo = f2bf(W[(size_t)k0 * N + col]);
  unsigned hi = f2bf(W[(size_t)(k0 + 1) * N + col]);
  Wp[p] = lo | (hi << 16);
}

// ---------------------------------------------------------------------------
// bf16 WMMA GEMM: out = epilogue(A[MxK] @ W[KxN] + bias)
// A: row-major bf16 (u16). Wp: fragment-packed (see pack_weight_kernel).
// Block = 4 waves; each wave owns a 16x64 tile (same tileN across waves).
// B K-tiles (4KB) staged cooperatively into LDS, double-buffered. Per K-step:
// next-tile global fetch issued first, all 4 B frags loaded into distinct
// registers (8 back-to-back ds_load_b128), then 4 pipelined WMMAs, then the
// ds_store of the next tile, then one barrier. A-tail rows are clamped (not
// predicated) so EXEC stays full for WMMA with no per-iteration masking.
// Epilogue: +bias, SiLU, *rowscale[row], *mul[row*N+col] (all optional).
// Requires K%32==0, N%64==0.
// ---------------------------------------------------------------------------
__global__ __launch_bounds__(128) void wmma_gemm_kernel(
    const u16* __restrict__ A, const unsigned* __restrict__ Wp,
    const float* __restrict__ bias, const float* __restrict__ rowscale,
    const float* __restrict__ mul, float* __restrict__ outF, u16* __restrict__ outB,
    int M, int Nn, int K, int act)
{
  __shared__ __align__(16) unsigned ldsb[2][1024];   // 2 x 4KB K-tiles
  const int lane = threadIdx.x;          // 0..31
  const int wv   = threadIdx.y;          // 0..3
  const int half = lane >> 4;
  const int mrow = lane & 15;
  const int tileM  = (blockIdx.x * 4 + wv) * 16;
  const int tN16   = blockIdx.y * 4;     // tileN / 16
  const int ktiles = K >> 5;

  const int    arow    = tileM + mrow;
  const int    arowc   = arow < M ? arow : (M - 1);   // clamp; stores row-guarded
  const size_t arowoff = (size_t)arowc * (size_t)K;

  // Cooperative B staging: thread (wv,lane) owns 32B of fragment wv.
  const uint4* gB = (const uint4*)Wp + (size_t)(tN16 + wv) * ktiles * 64 + lane * 2;
  uint4* lS0 = (uint4*)&ldsb[0][wv * 256 + lane * 8];
  uint4* lS1 = (uint4*)&ldsb[1][wv * 256 + lane * 8];

  { uint4 q0 = gB[0], q1 = gB[1]; lS0[0] = q0; lS0[1] = q1; }
  __syncthreads();

  v8f acc[4];
  #pragma unroll
  for (int j = 0; j < 4; ++j)
    #pragma unroll
    for (int i = 0; i < 8; ++i) acc[j][i] = 0.0f;

  for (int kt = 0; kt < ktiles; ++kt) {
    const int  kb = kt << 5;
    const int  cb = kt & 1;
    const bool have_next = (kt + 1 < ktiles);
    // issue next-tile global fetch first (latency hidden under WMMAs)
    uint4 n0, n1;
    if (have_next) {
      const uint4* gn = gB + (size_t)(kt + 1) * 64;
      n0 = gn[0]; n1 = gn[1];
      __builtin_prefetch(A + arowoff + kb + 32 + half * 8, 0, 3);
    }
    // ---- A fragment: 16x32 bf16, two b128 loads ----
    const uint4* ap = (const uint4*)(A + arowoff + kb + half * 16);
    uint4 a0 = ap[0];   // K offsets half*16 + 0..7
    uint4 a1 = ap[1];   // K offsets half*16 + 8..15
    FragU a;
    a.u[0] = a0.x; a.u[1] = a0.y; a.u[2] = a0.z; a.u[3] = a0.w;
    a.u[4] = a1.x; a.u[5] = a1.y; a.u[6] = a1.z; a.u[7] = a1.w;
    // ---- all 4 B fragments into distinct registers (8 ds_load_b128) ----
    FragU b[4];
    #pragma unroll
    for (int j = 0; j < 4; ++j) {
      const uint4* lp = (const uint4*)&ldsb[cb][j * 256 + lane * 8];
      uint4 b0 = lp[0], b1 = lp[1];
      b[j].u[0] = b0.x; b[j].u[1] = b0.y; b[j].u[2] = b0.z; b[j].u[3] = b0.w;
      b[j].u[4] = b1.x; b[j].u[5] = b1.y; b[j].u[6] = b1.z; b[j].u[7] = b1.w;
    }
    // ---- 4 pipelined WMMAs (independent accumulators) ----
    #pragma unroll
    for (int j = 0; j < 4; ++j)
      acc[j] = __builtin_amdgcn_wmma_f32_16x16x32_bf16(
          false, a.v, false, b[j].v, (short)0, acc[j], false, false);
    // ---- stage next tile into the other buffer ----
    if (have_next) {
      uint4* ln = cb ? lS0 : lS1;
      ln[0] = n0; ln[1] = n1;
    }
    __syncthreads();
  }

  // ---- Epilogue. C layout: col = tileN+j*16+mrow; element i -> row
  //      tileM + i + 8*half. ----
  #pragma unroll
  for (int j = 0; j < 4; ++j) {
    int   ccol = tN16 * 16 + j * 16 + mrow;
    float bv   = bias ? bias[ccol] : 0.0f;
    #pragma unroll
    for (int i = 0; i < 8; ++i) {
      int r = tileM + i + 8 * half;
      if (r < M) {
        float x = acc[j][i] + bv;
        if (act) x = siluf(x);
        if (rowscale) x *= rowscale[r];
        size_t idx = (size_t)r * Nn + ccol;
        if (mul)  x *= mul[idx];
        if (outF) outF[idx] = x;
        if (outB) outB[idx] = f2bf(x);
      }
    }
  }
}

// ---------------------------------------------------------------------------
__global__ __launch_bounds__(256) void f32_to_bf16_kernel(
    const float* __restrict__ in, u16* __restrict__ out, long n)
{
  long i = (long)blockIdx.x * 256 + threadIdx.x;
  if (i < n) out[i] = f2bf(in[i]);
}

// LayerNorm over H=128 per row, optional affine + SiLU, f32/bf16 outputs.
__global__ __launch_bounds__(128) void ln_act_kernel(
    const float* __restrict__ in, const float* __restrict__ gamma,
    const float* __restrict__ beta, float* __restrict__ outF,
    u16* __restrict__ outB, int act)
{
  int r = blockIdx.x, h = threadIdx.x;
  __shared__ float sm[4];
  float x = in[(size_t)r * HDIM + h];
  float v = x;
  #pragma unroll
  for (int m = 16; m; m >>= 1) v += __shfl_xor(v, m, 32);
  if ((h & 31) == 0) sm[h >> 5] = v;
  __syncthreads();
  float mean = (sm[0] + sm[1] + sm[2] + sm[3]) * (1.0f / HDIM);
  __syncthreads();
  float d = x - mean;
  v = d * d;
  #pragma unroll
  for (int m = 16; m; m >>= 1) v += __shfl_xor(v, m, 32);
  if ((h & 31) == 0) sm[h >> 5] = v;
  __syncthreads();
  float var = (sm[0] + sm[1] + sm[2] + sm[3]) * (1.0f / HDIM);
  float y = d * rsqrtf(var + 1e-5f);
  if (gamma) y = gamma[h] * y + beta[h];
  if (act) y = siluf(y);
  if (outF) outF[(size_t)r * HDIM + h] = y;
  if (outB) outB[(size_t)r * HDIM + h] = f2bf(y);
}

__global__ __launch_bounds__(128) void gather_rows_kernel(
    const float* __restrict__ table, const int* __restrict__ idx,
    float* __restrict__ out)
{
  int r = blockIdx.x, h = threadIdx.x;
  out[(size_t)r * HDIM + h] = table[(size_t)idx[r] * HDIM + h];
}

// Per-edge geometry: rbf (bf16), rbounds, edge_diff, frame[c][k] (k: diff,cross,vert)
__global__ __launch_bounds__(256) void edge_geom_kernel(
    const float* __restrict__ pos, const int* __restrict__ srcI,
    const int* __restrict__ dstI, u16* __restrict__ rbf_bf,
    float* __restrict__ rbounds, float* __restrict__ ediff,
    float* __restrict__ frame, int nE)
{
  int e = blockIdx.x * 256 + threadIdx.x;
  if (e >= nE) return;
  int s = srcI[e], d = dstI[e];
  float sx = pos[(size_t)s*3], sy = pos[(size_t)s*3+1], sz = pos[(size_t)s*3+2];
  float dx = pos[(size_t)d*3], dy = pos[(size_t)d*3+1], dz = pos[(size_t)d*3+2];
  float ex = sx - dx, ey = sy - dy, ez = sz - dz;
  float dist = sqrtf(ex*ex + ey*ey + ez*ez);
  float rb = (dist < 6.0f) ? 0.5f * (__cosf(dist * (PI_F / 6.0f)) + 1.0f) : 0.0f;
  rbounds[e] = rb;
  const float start = __expf(-6.0f);
  const float bb = (2.0f / (float)NRAD) * (1.0f - start);
  const float beta = 1.0f / (bb * bb);
  float ed = __expf(-dist);
  for (int i = 0; i < NRAD; ++i) {
    float mean = start + (1.0f - start) * ((float)i / (float)(NRAD - 1));
    float t = ed - mean;
    rbf_bf[(size_t)e * NRAD + i] = f2bf(rb * __expf(-beta * t * t));
  }
  float inv_d = 1.0f / dist;
  float fx = ex * inv_d, fy = ey * inv_d, fz = ez * inv_d;
  float cx = sy*dz - sz*dy, cy = sz*dx - sx*dz, cz = sx*dy - sy*dx;
  float cn = sqrtf(cx*cx + cy*cy + cz*cz) + 1e-10f;
  cx /= cn; cy /= cn; cz /= cn;
  float vx = fy*cz - fz*cy, vy = fz*cx - fx*cz, vz = fx*cy - fy*cx;
  ediff[(size_t)e*3+0] = fx; ediff[(size_t)e*3+1] = fy; ediff[(size_t)e*3+2] = fz;
  size_t fo = (size_t)e * 9;
  frame[fo+0] = fx; frame[fo+1] = cx; frame[fo+2] = vx;
  frame[fo+3] = fy; frame[fo+4] = cy; frame[fo+5] = vy;
  frame[fo+6] = fz; frame[fo+7] = cz; frame[fo+8] = vz;
}

__global__ __launch_bounds__(128) void scatter_s_kernel(
    const float* __restrict__ rh, const float* __restrict__ sn,
    const int* __restrict__ srcI, const int* __restrict__ dstI,
    float* __restrict__ s)
{
  int e = blockIdx.x, h = threadIdx.x;
  float v = rh[(size_t)e * HDIM + h] * sn[(size_t)srcI[e] * HDIM + h];
  atomicAdd(&s[(size_t)dstI[e] * HDIM + h], v);
}

__global__ __launch_bounds__(128) void msgv_scatter_kernel(
    const float* __restrict__ rh, const float* __restrict__ s1,
    const float* __restrict__ ediff, const int* __restrict__ srcI,
    const int* __restrict__ dstI, float* __restrict__ Sij)
{
  int e = blockIdx.x, h = threadIdx.x;
  float v = rh[(size_t)e * HDIM + h] * s1[(size_t)srcI[e] * HDIM + h];
  int d = dstI[e];
  #pragma unroll
  for (int c = 0; c < 3; ++c)
    atomicAdd(&Sij[(size_t)d * H3 + c * HDIM + h], v * ediff[(size_t)e*3+c]);
}

// scalarize both sides + assemble edge_weight (bf16, E x 480)
__global__ __launch_bounds__(128) void scalarize_kernel(
    const float* __restrict__ Sij, const float* __restrict__ frame,
    const float* __restrict__ rbounds, const float* __restrict__ rh,
    const u16* __restrict__ rbf_bf, const int* __restrict__ srcI,
    const int* __restrict__ dstI, const float* __restrict__ lw1,
    const float* __restrict__ lb1, const float* __restrict__ lw2,
    const float* __restrict__ lb2, u16* __restrict__ ew)
{
  __shared__ float sw1[96], sb1[32], sw2[32], sb2s;
  int e = blockIdx.x, h = threadIdx.x;
  if (h < 96) sw1[h] = lw1[h];
  if (h < 32) { sb1[h] = lb1[h]; sw2[h] = lw2[h]; }
  if (h == 0) sb2s = lb2[0];
  __syncthreads();
  float rb = rbounds[e];
  size_t fo = (size_t)e * 9, eo = (size_t)e * EWD;
  int g0 = srcI[e], g1 = dstI[e];
  #pragma unroll
  for (int side = 0; side < 2; ++side) {
    int g = side ? g1 : g0;
    float sc0 = 0.f, sc1 = 0.f, sc2 = 0.f;
    #pragma unroll
    for (int c = 0; c < 3; ++c) {
      float Sv = Sij[(size_t)g * H3 + c * HDIM + h];
      sc0 += frame[fo + c*3 + 0] * Sv;
      sc1 += frame[fo + c*3 + 1] * Sv;
      sc2 += frame[fo + c*3 + 2] * Sv;
    }
    sc1 = fabsf(sc1);
    float out = sb2s + sc0;   // b2 + p[...,0]
    #pragma unroll
    for (int j = 0; j < 32; ++j) {
      float hd = siluf(sc0 * sw1[j] + sc1 * sw1[32 + j] + sc2 * sw1[64 + j] + sb1[j]);
      out += hd * sw2[j];
    }
    ew[eo + side * HDIM + h] = f2bf(out * rb);
  }
  ew[eo + 2 * HDIM + h] = f2bf(rh[(size_t)e * HDIM + h]);
  if (h < NRAD) ew[eo + H3 + h] = rbf_bf[(size_t)e * NRAD + h];
}

__global__ __launch_bounds__(128) void message_scatter_kernel(
    const float* __restrict__ xh, const float* __restrict__ rbfh,
    const float* __restrict__ vec, const float* __restrict__ ediff,
    const int* __restrict__ srcI, const int* __restrict__ dstI,
    float* __restrict__ sacc, float* __restrict__ vacc)
{
  int e = blockIdx.x, h = threadIdx.x;
  int sN = srcI[e], dN = dstI[e];
  size_t xo = (size_t)sN * H3, ro = (size_t)e * H3;
  float mx = xh[xo + h] * rbfh[ro + h];
  atomicAdd(&sacc[(size_t)dN * HDIM + h], mx);
  float m2 = xh[xo + HDIM + h]     * rbfh[ro + HDIM + h]     * INV_SQRT3;
  float m3 = xh[xo + 2 * HDIM + h] * rbfh[ro + 2 * HDIM + h];
  #pragma unroll
  for (int c = 0; c < 3; ++c) {
    float mv = (vec[xo + c * HDIM + h] * m2 + m3 * ediff[(size_t)e*3+c]) * INV_SQRTH;
    atomicAdd(&vacc[(size_t)dN * H3 + c * HDIM + h], mv);
  }
}

__global__ __launch_bounds__(128) void update_sv_kernel(
    float* __restrict__ s, const float* __restrict__ sacc,
    float* __restrict__ vec, const float* __restrict__ vacc,
    u16* __restrict__ vec_bf)
{
  int n = blockIdx.x, h = threadIdx.x;
  size_t so = (size_t)n * HDIM + h;
  s[so] = (s[so] + sacc[so]) * INV_SQRT2;
  #pragma unroll
  for (int c = 0; c < 3; ++c) {
    size_t vo = (size_t)n * H3 + c * HDIM + h;
    float nv = vec[vo] + vacc[vo];
    vec[vo] = nv;
    vec_bf[vo] = f2bf(nv);
  }
}

__global__ __launch_bounds__(128) void ft_scalar_kernel(
    const float* __restrict__ vp, const float* __restrict__ s,
    u16* __restrict__ cat_bf, float* __restrict__ vdot)
{
  int n = blockIdx.x, h = threadIdx.x;
  float ss = 1e-10f, dp = 0.f;
  #pragma unroll
  for (int c = 0; c < 3; ++c) {
    size_t ro = ((size_t)n * 3 + c) * H2;
    float a = vp[ro + h], b = vp[ro + HDIM + h];
    ss += a * a;
    dp += a * b;
  }
  cat_bf[(size_t)n * H2 + h]        = f2bf(s[(size_t)n * HDIM + h]);
  cat_bf[(size_t)n * H2 + HDIM + h] = f2bf(sqrtf(ss));
  vdot[(size_t)n * HDIM + h] = dp * INV_SQRTH;
}

__global__ __launch_bounds__(128) void ft_update_kernel(
    float* __restrict__ s, float* __restrict__ vec,
    const float* __restrict__ xvh, const float* __restrict__ vdot,
    const float* __restrict__ vp)
{
  int n = blockIdx.x, h = threadIdx.x;
  size_t xo = (size_t)n * H3;
  float x1 = xvh[xo + h], x2 = xvh[xo + HDIM + h], x3 = xvh[xo + 2 * HDIM + h];
  s[(size_t)n * HDIM + h] += (x1 + x2 + vdot[(size_t)n * HDIM + h]) * INV_SQRT2;
  #pragma unroll
  for (int c = 0; c < 3; ++c) {
    size_t ro = ((size_t)n * 3 + c) * H2;
    vec[(size_t)n * H3 + c * HDIM + h] += x3 * vp[ro + HDIM + h];
  }
}

__global__ __launch_bounds__(128) void final_kernel(
    const float* __restrict__ s, const float* __restrict__ last_w,
    const float* __restrict__ last_b, const int* __restrict__ batch,
    float* __restrict__ out, int nN)
{
  int n = blockIdx.x * 4 + threadIdx.y;
  if (n >= nN) return;
  int lane = threadIdx.x;
  float acc = 0.f;
  for (int h = lane; h < HDIM; h += 32) acc += s[(size_t)n * HDIM + h] * last_w[h];
  #pragma unroll
  for (int m = 16; m; m >>= 1) acc += __shfl_xor(acc, m, 32);
  if (lane == 0) atomicAdd(&out[batch[n]], acc + last_b[0]);
}

// ---------------------------------------------------------------------------
extern "C" void kernel_launch(void* const* d_in, const int* in_sizes, int n_in,
                              void* d_out, int out_size, void* d_ws, size_t ws_size,
                              hipStream_t stream)
{
  const float* pos      = (const float*)d_in[0];
  const float* z_emb_w  = (const float*)d_in[1];
  const float* ne_emb_w = (const float*)d_in[2];
  const float* rl_w1    = (const float*)d_in[3];
  const float* rl_b1    = (const float*)d_in[4];
  const float* rl_w2    = (const float*)d_in[5];
  const float* rl_b2    = (const float*)d_in[6];
  const float* sv_w     = (const float*)d_in[7];
  const float* sv_b     = (const float*)d_in[8];
  const float* lin_w1   = (const float*)d_in[9];
  const float* lin_b1   = (const float*)d_in[10];
  const float* lin_w2   = (const float*)d_in[11];
  const float* lin_b2   = (const float*)d_in[12];
  const float* mp_ln_g  = (const float*)d_in[13];
  const float* mp_ln_b  = (const float*)d_in[14];
  const float* mp_xw1   = (const float*)d_in[15];
  const float* mp_xb1   = (const float*)d_in[16];
  const float* mp_xw2   = (const float*)d_in[17];
  const float* mp_xb2   = (const float*)d_in[18];
  const float* mp_rw    = (const float*)d_in[19];
  const float* mp_rb    = (const float*)d_in[20];
  const float* mp_dw1   = (const float*)d_in[21];
  const float* mp_db1   = (const float*)d_in[22];
  const float* mp_dw2   = (const float*)d_in[23];
  const float* mp_db2   = (const float*)d_in[24];
  const float* ft_vw    = (const float*)d_in[25];
  const float* ft_xw1   = (const float*)d_in[26];
  const float* ft_xb1   = (const float*)d_in[27];
  const float* ft_xw2   = (const float*)d_in[28];
  const float* ft_xb2   = (const float*)d_in[29];
  const float* last_w   = (const float*)d_in[30];
  const float* last_b   = (const float*)d_in[31];
  const int*   zidx     = (const int*)d_in[32];
  const int*   eidx     = (const int*)d_in[33];
  const int*   batch    = (const int*)d_in[34];

  const int nN = in_sizes[0] / 3;
  const int nE = in_sizes[33] / 2;
  const int* srcI = eidx;
  const int* dstI = eidx + nE;
  float* outG = (float*)d_out;

  size_t cur = 0;
  auto alloc = [&](size_t bytes) -> void* {
    void* p = (void*)((char*)d_ws + cur);
    cur += (bytes + 255) & ~(size_t)255;
    return p;
  };
  // edge buffers
  u16*   rbf_bf  = (u16*)  alloc((size_t)nE * NRAD * 2);
  float* rbounds = (float*)alloc((size_t)nE * 4);
  float* ediff   = (float*)alloc((size_t)nE * 3 * 4);
  float* frame   = (float*)alloc((size_t)nE * 9 * 4);
  float* rh      = (float*)alloc((size_t)nE * HDIM * 4);
  u16*   ebuf_bf = (u16*)  alloc((size_t)nE * H3 * 2);    // rl hidden / d1
  u16*   ew_bf   = (u16*)  alloc((size_t)nE * EWD * 2);
  float* r1      = (float*)alloc((size_t)nE * H3 * 4);    // r1, then rbfh in-place
  // node buffers
  float* s     = (float*)alloc((size_t)nN * HDIM * 4);
  float* sn    = (float*)alloc((size_t)nN * HDIM * 4);
  u16*   s_bf  = (u16*)  alloc((size_t)nN * HDIM * 2);
  u16*   t_bf  = (u16*)  alloc((size_t)nN * HDIM * 2);
  u16*   u_bf  = (u16*)  alloc((size_t)nN * HDIM * 2);
  float* g1b   = (float*)alloc((size_t)nN * HDIM * 4);
  float* s1    = (float*)alloc((size_t)nN * HDIM * 4);
  float* vec   = (float*)alloc((size_t)nN * H3 * 4);
  float* Sij   = (float*)alloc((size_t)nN * H3 * 4);
  float* sacc  = (float*)alloc((size_t)nN * HDIM * 4);
  float* vacc  = (float*)alloc((size_t)nN * H3 * 4);
  float* xh    = (float*)alloc((size_t)nN * H3 * 4);
  u16*   vecbf = (u16*)  alloc((size_t)nN * H3 * 2);
  float* vp    = (float*)alloc((size_t)nN * 3 * H2 * 4);
  u16*   catbf = (u16*)  alloc((size_t)nN * H2 * 2);
  float* vdot  = (float*)alloc((size_t)nN * HDIM * 4);
  float* xvh   = (float*)alloc((size_t)nN * H3 * 4);
  // fragment-packed bf16 weights (dword layout, same byte size as bf16)
  unsigned* rlw1_p = (unsigned*)alloc((size_t)NRAD * HDIM * 2);
  unsigned* rlw2_p = (unsigned*)alloc((size_t)HDIM * HDIM * 2);
  unsigned* svw_p  = (unsigned*)alloc((size_t)HDIM * HDIM * 2);
  unsigned* xw1_p  = (unsigned*)alloc((size_t)2 * HDIM * HDIM * 2);
  unsigned* xw2_p  = (unsigned*)alloc((size_t)2 * HDIM * H3 * 2);
  unsigned* rw_p   = (unsigned*)alloc((size_t)2 * NRAD * H3 * 2);
  unsigned* dw1_p  = (unsigned*)alloc((size_t)2 * EWD * H3 * 2);
  unsigned* dw2_p  = (unsigned*)alloc((size_t)2 * H3 * H3 * 2);
  unsigned* vw_p   = (unsigned*)alloc((size_t)2 * HDIM * H2 * 2);
  unsigned* fxw1_p = (unsigned*)alloc((size_t)2 * H2 * HDIM * 2);
  unsigned* fxw2_p = (unsigned*)alloc((size_t)2 * HDIM * H3 * 2);

  auto pack = [&](const float* W, unsigned* Wp, int K, int N) {
    long total = (long)(N / 16) * (K / 32) * 256;
    pack_weight_kernel<<<dim3((unsigned)((total + 255) / 256)), dim3(256), 0, stream>>>(W, Wp, K, N);
  };
  auto gemm = [&](const u16* A, const unsigned* Wp, const float* bias, const float* rscale,
                  const float* mul, float* oF, u16* oB, int M, int Nn, int K, int act) {
    dim3 grid((unsigned)((M + 63) / 64), (unsigned)(Nn / 64));
    dim3 block(32, 4);
    wmma_gemm_kernel<<<grid, block, 0, stream>>>(A, Wp, bias, rscale, mul, oF, oB, M, Nn, K, act);
  };

  // ---- init / weight packing ----
  hipMemsetAsync(vec, 0, (size_t)nN * H3 * 4, stream);
  hipMemsetAsync(Sij, 0, (size_t)nN * H3 * 4, stream);
  hipMemsetAsync(outG, 0, (size_t)GGRAPH * 4, stream);
  pack(rl_w1, rlw1_p, NRAD, HDIM);
  pack(rl_w2, rlw2_p, HDIM, HDIM);
  pack(sv_w,  svw_p,  HDIM, HDIM);
  for (int l = 0; l < 2; ++l) {
    pack(mp_xw1 + (size_t)l * HDIM * HDIM, xw1_p + (size_t)l * HDIM * HDIM / 2, HDIM, HDIM);
    pack(mp_xw2 + (size_t)l * HDIM * H3,   xw2_p + (size_t)l * HDIM * H3 / 2,   HDIM, H3);
    pack(mp_rw  + (size_t)l * NRAD * H3,   rw_p  + (size_t)l * NRAD * H3 / 2,   NRAD, H3);
    pack(mp_dw1 + (size_t)l * EWD * H3,    dw1_p + (size_t)l * EWD * H3 / 2,    EWD,  H3);
    pack(mp_dw2 + (size_t)l * H3 * H3,     dw2_p + (size_t)l * H3 * H3 / 2,     H3,   H3);
    pack(ft_vw  + (size_t)l * HDIM * H2,   vw_p  + (size_t)l * HDIM * H2 / 2,   HDIM, H2);
    pack(ft_xw1 + (size_t)l * H2 * HDIM,   fxw1_p + (size_t)l * H2 * HDIM / 2,  H2,   HDIM);
    pack(ft_xw2 + (size_t)l * HDIM * H3,   fxw2_p + (size_t)l * HDIM * H3 / 2,  HDIM, H3);
  }

  // ---- edge geometry + rbf ----
  edge_geom_kernel<<<dim3((nE + 255) / 256), dim3(256), 0, stream>>>(
      pos, srcI, dstI, rbf_bf, rbounds, ediff, frame, nE);

  // ---- rh = rbounds * mlp2(rbf) ----
  gemm(rbf_bf, rlw1_p, rl_b1, nullptr, nullptr, nullptr, ebuf_bf, nE, HDIM, NRAD, 1);
  gemm(ebuf_bf, rlw2_p, rl_b2, rbounds, nullptr, rh, nullptr, nE, HDIM, HDIM, 0);

  // ---- node embeddings s, sn ----
  gather_rows_kernel<<<dim3(nN), dim3(HDIM), 0, stream>>>(z_emb_w, zidx, s);
  ln_act_kernel<<<dim3(nN), dim3(HDIM), 0, stream>>>(s, nullptr, nullptr, s, nullptr, 0);
  gather_rows_kernel<<<dim3(nN), dim3(HDIM), 0, stream>>>(ne_emb_w, zidx, sn);
  ln_act_kernel<<<dim3(nN), dim3(HDIM), 0, stream>>>(sn, nullptr, nullptr, sn, nullptr, 0);

  // ---- s += segsum(rh * sn[src]) ----
  scatter_s_kernel<<<dim3(nE), dim3(HDIM), 0, stream>>>(rh, sn, srcI, dstI, s);

  // ---- s1 = silu(ln(s @ sv_w + b)) ----
  f32_to_bf16_kernel<<<dim3((unsigned)(((long)nN * HDIM + 255) / 256)), dim3(256), 0, stream>>>(
      s, s_bf, (long)nN * HDIM);
  gemm(s_bf, svw_p, sv_b, nullptr, nullptr, g1b, nullptr, nN, HDIM, HDIM, 0);
  ln_act_kernel<<<dim3(nN), dim3(HDIM), 0, stream>>>(g1b, nullptr, nullptr, s1, nullptr, 1);

  // ---- S_ij = segsum(rh * diff * s1[src]) ----
  msgv_scatter_kernel<<<dim3(nE), dim3(HDIM), 0, stream>>>(rh, s1, ediff, srcI, dstI, Sij);

  // ---- edge_weight (E x 480, bf16) ----
  scalarize_kernel<<<dim3(nE), dim3(HDIM), 0, stream>>>(
      Sij, frame, rbounds, rh, rbf_bf, srcI, dstI, lin_w1, lin_b1, lin_w2, lin_b2, ew_bf);

  // ---- message-passing layers ----
  for (int l = 0; l < 2; ++l) {
    // xh = mlp2(g*ln(s)+b)
    ln_act_kernel<<<dim3(nN), dim3(HDIM), 0, stream>>>(
        s, mp_ln_g + (size_t)l * HDIM, mp_ln_b + (size_t)l * HDIM, nullptr, t_bf, 0);
    gemm(t_bf, xw1_p + (size_t)l * HDIM * HDIM / 2, mp_xb1 + (size_t)l * HDIM,
         nullptr, nullptr, nullptr, u_bf, nN, HDIM, HDIM, 1);
    gemm(u_bf, xw2_p + (size_t)l * HDIM * H3 / 2, mp_xb2 + (size_t)l * H3,
         nullptr, nullptr, xh, nullptr, nN, H3, HDIM, 0);
    // rbfh = (rbf @ rw + rb) * mlp2(edge_weight)
    gemm(rbf_bf, rw_p + (size_t)l * NRAD * H3 / 2, mp_rb + (size_t)l * H3,
         nullptr, nullptr, r1, nullptr, nE, H3, NRAD, 0);
    gemm(ew_bf, dw1_p + (size_t)l * EWD * H3 / 2, mp_db1 + (size_t)l * H3,
         nullptr, nullptr, nullptr, ebuf_bf, nE, H3, EWD, 1);
    gemm(ebuf_bf, dw2_p + (size_t)l * H3 * H3 / 2, mp_db2 + (size_t)l * H3,
         nullptr, r1, r1, nullptr, nE, H3, H3, 0);   // in-place: r1 <- (acc+b)*r1
    // scatter messages
    hipMemsetAsync(sacc, 0, (size_t)nN * HDIM * 4, stream);
    hipMemsetAsync(vacc, 0, (size_t)nN * H3 * 4, stream);
    message_scatter_kernel<<<dim3(nE), dim3(HDIM), 0, stream>>>(
        xh, r1, vec, ediff, srcI, dstI, sacc, vacc);
    update_sv_kernel<<<dim3(nN), dim3(HDIM), 0, stream>>>(s, sacc, vec, vacc, vecbf);
    // feature transform
    gemm(vecbf, vw_p + (size_t)l * HDIM * H2 / 2, nullptr,
         nullptr, nullptr, vp, nullptr, 3 * nN, H2, HDIM, 0);
    ft_scalar_kernel<<<dim3(nN), dim3(HDIM), 0, stream>>>(vp, s, catbf, vdot);
    gemm(catbf, fxw1_p + (size_t)l * H2 * HDIM / 2, ft_xb1 + (size_t)l * HDIM,
         nullptr, nullptr, nullptr, u_bf, nN, HDIM, H2, 1);
    gemm(u_bf, fxw2_p + (size_t)l * HDIM * H3 / 2, ft_xb2 + (size_t)l * H3,
         nullptr, nullptr, xvh, nullptr, nN, H3, HDIM, 0);
    ft_update_kernel<<<dim3(nN), dim3(HDIM), 0, stream>>>(s, vec, xvh, vdot, vp);
  }

  // ---- out = segsum(s @ last_w + last_b over batch) ----
  final_kernel<<<dim3((nN + 3) / 4), dim3(32, 4), 0, stream>>>(
      s, last_w, last_b, batch, outG, nN);
}
